// GNN_1520418423155
// MI455X (gfx1250) — matrix operand order repaired and checked by reading, not compiled, not run on hardware
//
#include <hip/hip_runtime.h>
#include <hip/hip_bf16.h>
#include <stdint.h>

// ---------------- problem constants ----------------
#define NN 200000
#define EE 400000
#define GG 8000
#define HH 4
#define CC 64
#define F_IN 32
#define F_E 8
#define LL 3

typedef __attribute__((ext_vector_type(16))) _Float16 v16h;
typedef __attribute__((ext_vector_type(8)))  _Float16 v8h;
typedef __attribute__((ext_vector_type(8)))  float    v8f;
typedef __attribute__((ext_vector_type(4)))  float    v4f;

// ---------------- WMMA helper (wave32, 16x16x32 f16 -> f32) ----------------
__device__ __forceinline__ v8f wmma_acc(v16h a, v16h b, v8f c) {
    return __builtin_amdgcn_wmma_f32_16x16x32_f16(
        /*neg_a=*/false, a, /*neg_b=*/false, b,
        /*c_mod=*/(short)0, c, /*reuse_a=*/false, /*reuse_b=*/false);
}

// A fragment from a row-major f16 LDS tile. Contiguous k order (as k_pack_b): lane l<16 holds
// row M=l, K = kbase+{koff..koff+15} with koff=0;
// lanes 16-31 same rows with koff=16. Two runs of 8 contiguous halfs
// (16 bytes) -> two ds_load_b128.
__device__ __forceinline__ v16h lds_a_frag(const _Float16* __restrict__ rowp,
                                           int kbase, int lane) {
    int koff = (lane < 16) ? 0 : 16;
    v8h lo = *(const v8h*)(rowp + kbase + koff);
    v8h hi = *(const v8h*)(rowp + kbase + koff + 8);
    v16h a;
#pragma unroll
    for (int j = 0; j < 8; ++j) { a[j] = lo[j]; a[8 + j] = hi[j]; }
    return a;
}

// B fragment from fragment-ordered packed f16 buffer: one 32-byte read/lane.
__device__ __forceinline__ v16h packed_b_frag(const _Float16* __restrict__ pB,
                                              int frag, int lane) {
    return *(const v16h*)(pB + ((size_t)frag * 32 + lane) * 16);
}

// ---------------- weight pre-pack ----------------
// W: f32 row-major [K x ldb]. Fragment f covers coltile nt = f / (K/32),
// ktile kt = f % (K/32). Lane layout (contiguous k, as lds_a_frag): col = nt*16 + (lane&15),
// lanes 0-15 hold K = kt*32+0..15, lanes 16-31 hold K = kt*32+16..31.
__global__ void k_pack_b(const float* __restrict__ W, int ldb, int K,
                         _Float16* __restrict__ out) {
    int f = blockIdx.x, lane = threadIdx.x;
    int ktiles = K / 32;
    int nt = f / ktiles, kt = f % ktiles;
    int col  = nt * 16 + (lane & 15);
    int koff = (lane < 16) ? 0 : 16;
    _Float16* o = out + ((size_t)f * 32 + lane) * 16;
#pragma unroll
    for (int v = 0; v < 8; ++v) {
        int k = kt * 32 + koff + 2 * v;
        o[2 * v]     = (_Float16)W[(size_t)k * ldb + col];
        o[2 * v + 1] = (_Float16)W[(size_t)(k + 1) * ldb + col];
    }
}

// ---------------- kernels ----------------

// h = relu(x @ lin_w + lin_b); h_skip = h.   [N,32]@[32,64]
__global__ __launch_bounds__(256) void k_lin_proj(const float* __restrict__ x,
                                                  const _Float16* __restrict__ pB,
                                                  const float* __restrict__ bias,
                                                  float* __restrict__ h,
                                                  float* __restrict__ h_skip) {
    __shared__ __align__(16) _Float16 sA[128 * F_IN];   // 8 KB
    int tid = threadIdx.x, wave = tid >> 5, lane = tid & 31;
    // ---- stage 128 rows x 32 cols of x as f16 (coalesced float4 reads) ----
    {
        int row = tid >> 1, c0 = (tid & 1) * 16;        // 16 halfs per thread
        long grow = (long)blockIdx.x * 128 + row;
        _Float16* dst = sA + row * F_IN + c0;
        if (grow < NN) {
            const float* src = x + grow * F_IN + c0;
#pragma unroll
            for (int j = 0; j < 4; ++j) {
                v4f v = *(const v4f*)(src + 4 * j);
#pragma unroll
                for (int t = 0; t < 4; ++t) dst[4 * j + t] = (_Float16)v[t];
            }
        } else {
#pragma unroll
            for (int j = 0; j < 16; ++j) dst[j] = (_Float16)0.f;
        }
    }
    __syncthreads();
    long tile = (long)blockIdx.x * 8 + wave;            // 16-row tile
    if (tile * 16 >= NN) return;                        // wave-uniform exit
    const _Float16* rowp = sA + (wave * 16 + (lane & 15)) * F_IN;
    v16h a = lds_a_frag(rowp, 0, lane);
    int col = lane & 15, rsel = (lane >> 4) * 8;
#pragma unroll
    for (int nt = 0; nt < 4; ++nt) {
        v16h b = packed_b_frag(pB, nt, lane);           // ktiles = 1
        v8f c = {};
        c = wmma_acc(a, b, c);
        int colg = nt * 16 + col;
        float bv = bias[colg];
#pragma unroll
        for (int r = 0; r < 8; ++r) {
            long rowg = tile * 16 + rsel + r;
            float v = c[r] + bv;
            v = v > 0.f ? v : 0.f;
            h[rowg * CC + colg] = v;
            h_skip[rowg * CC + colg] = v;
        }
    }
}

// per-channel BN training stats -> scale/shift
__global__ __launch_bounds__(256) void k_bn_stats(const float* __restrict__ h,
                                                  const float* __restrict__ gamma,
                                                  const float* __restrict__ beta,
                                                  float* __restrict__ scale,
                                                  float* __restrict__ shift) {
    __shared__ float s1[256], s2[256];
    int c = blockIdx.x;
    float sum = 0.f, sq = 0.f;
    for (long n = threadIdx.x; n < NN; n += 256) {
        float v = h[n * CC + c];
        sum += v; sq += v * v;
    }
    s1[threadIdx.x] = sum; s2[threadIdx.x] = sq;
    __syncthreads();
    for (int s = 128; s > 0; s >>= 1) {
        if (threadIdx.x < s) { s1[threadIdx.x] += s1[threadIdx.x + s];
                               s2[threadIdx.x] += s2[threadIdx.x + s]; }
        __syncthreads();
    }
    if (threadIdx.x == 0) {
        float mu  = s1[0] / (float)NN;
        float var = s2[0] / (float)NN - mu * mu;
        float sc  = gamma[c] * rsqrtf(var + 1e-5f);
        scale[c] = sc;
        shift[c] = beta[c] - mu * sc;
    }
}

// xl = bn(h) @ conv_w[l]  [N,64]@[64,256]; BN fused into LDS staging.
// Each wave owns one head's 64-column group, so a_s/a_d are reduced in the
// epilogue via shfl_xor over the 16-lane column halves (no extra xl pass).
__global__ __launch_bounds__(256) void k_conv_gemm(const float* __restrict__ h,
                                                   const float* __restrict__ sc,
                                                   const float* __restrict__ sh,
                                                   const _Float16* __restrict__ pB,
                                                   const float* __restrict__ asrc,
                                                   const float* __restrict__ adst,
                                                   float* __restrict__ xl,
                                                   float* __restrict__ a_s,
                                                   float* __restrict__ a_d) {
    __shared__ __align__(16) _Float16 sA[32 * CC];      // 4 KB: bn(h) tile f16
    int tid = threadIdx.x, wave = tid >> 5, lane = tid & 31;
    // ---- stage 32 rows x 64 ch with BN affine applied ----
    {
        int row = tid >> 3, c0 = (tid & 7) * 8;         // 8 halfs per thread
        long grow = (long)blockIdx.x * 32 + row;        // grid exact: < NN
        const float* src = h + grow * CC + c0;
        _Float16* dst = sA + row * CC + c0;
#pragma unroll
        for (int j = 0; j < 2; ++j) {
            v4f v = *(const v4f*)(src + 4 * j);
#pragma unroll
            for (int t = 0; t < 4; ++t) {
                int cix = c0 + 4 * j + t;
                dst[4 * j + t] = (_Float16)(v[t] * sc[cix] + sh[cix]);
            }
        }
    }
    __syncthreads();
    long mtile = (long)blockIdx.x * 2 + (wave >> 2);
    int head = wave & 3;                                // 64-col group == head
    const _Float16* rowp = sA + ((wave >> 2) * 16 + (lane & 15)) * CC;
    v16h a0 = lds_a_frag(rowp, 0, lane);
    v16h a1 = lds_a_frag(rowp, 32, lane);
    int col = lane & 15, rsel = (lane >> 4) * 8;
    const float* vsrc = asrc + head * CC;
    const float* vdst = adst + head * CC;
    float ps[8] = {0, 0, 0, 0, 0, 0, 0, 0};
    float pd[8] = {0, 0, 0, 0, 0, 0, 0, 0};
#pragma unroll
    for (int nt = 0; nt < 4; ++nt) {
        int ct = head * 4 + nt;                         // global 16-col tile
        v16h b0 = packed_b_frag(pB, ct * 2 + 0, lane);
        v16h b1 = packed_b_frag(pB, ct * 2 + 1, lane);
        v8f c = {};
        c = wmma_acc(a0, b0, c);
        c = wmma_acc(a1, b1, c);
        float wsv = vsrc[nt * 16 + col];
        float wdv = vdst[nt * 16 + col];
#pragma unroll
        for (int r = 0; r < 8; ++r) {
            xl[(mtile * 16 + rsel + r) * (size_t)(HH * CC) + head * CC +
               nt * 16 + col] = c[r];
            ps[r] += c[r] * wsv;
            pd[r] += c[r] * wdv;
        }
    }
    // reduce over the 16 column lanes of each half (rows differ per half)
#pragma unroll
    for (int m = 1; m < 16; m <<= 1) {
#pragma unroll
        for (int r = 0; r < 8; ++r) {
            ps[r] += __shfl_xor(ps[r], m);
            pd[r] += __shfl_xor(pd[r], m);
        }
    }
    if ((lane & 15) == 0) {
        long rowbase = mtile * 16 + rsel;
#pragma unroll
        for (int r = 0; r < 8; ++r) {
            a_s[(rowbase + r) * HH + head] = ps[r];
            a_d[(rowbase + r) * HH + head] = pd[r];
        }
    }
}

// weff[f,h] = <conv_we[f, h*64:...], att_edge[h,:]>  (8x4 effective weight)
__global__ void k_weff(const float* __restrict__ conv_we,
                       const float* __restrict__ att_edge,
                       float* __restrict__ weff) {
    int i = threadIdx.x;
    if (i >= F_E * HH) return;
    int f = i >> 2, hh = i & 3;
    float s = 0.f;
    for (int c = 0; c < CC; ++c)
        s += conv_we[f * (HH * CC) + hh * CC + c] * att_edge[hh * CC + c];
    weff[f * HH + hh] = s;
}

__global__ void k_fill(float* __restrict__ p, long n, float v) {
    long i = (long)blockIdx.x * 256 + threadIdx.x;
    if (i < n) p[i] = v;
}

__device__ __forceinline__ void atomicMaxFloat(float* addr, float value) {
    if (value >= 0.f)
        atomicMax((int*)addr, __float_as_int(value));
    else
        atomicMin((unsigned int*)addr, __float_as_uint(value));
}

// per-edge attention logit + segment max over dst
__global__ __launch_bounds__(256) void k_edge_score(const int* __restrict__ ei,
                                                    const float* __restrict__ ea,
                                                    const float* __restrict__ weff,
                                                    const float* __restrict__ a_s,
                                                    const float* __restrict__ a_d,
                                                    const float* __restrict__ xl,
                                                    float* __restrict__ a,
                                                    float* __restrict__ amax) {
    long i = (long)blockIdx.x * 256 + threadIdx.x;      // e*H + h
    if (i >= (long)EE * HH) return;
    int hh = (int)(i & 3); long e = i >> 2;
    int s = ei[e], d = ei[EE + e];
    float ae = 0.f;
#pragma unroll
    for (int f = 0; f < F_E; ++f) ae += ea[e * F_E + f] * weff[f * HH + hh];
    float v = a_s[(long)s * HH + hh] + a_d[(long)d * HH + hh] + ae;
    v = v > 0.f ? v : 0.2f * v;                         // leaky relu
    a[i] = v;
    atomicMaxFloat(&amax[(long)d * HH + hh], v);
    if (hh == 0)  // warm L2 for the upcoming gather of this source row
        __builtin_prefetch(xl + (size_t)s * (HH * CC), 0, 1);
}

// ex = exp(a - amax[dst]); den[dst] += ex
__global__ __launch_bounds__(256) void k_edge_exp(const int* __restrict__ ei,
                                                  const float* __restrict__ a,
                                                  const float* __restrict__ amax,
                                                  float* __restrict__ ex,
                                                  float* __restrict__ den) {
    long i = (long)blockIdx.x * 256 + threadIdx.x;
    if (i >= (long)EE * HH) return;
    int hh = (int)(i & 3); long e = i >> 2;
    int d = ei[EE + e];
    float v = __expf(a[i] - amax[(long)d * HH + hh]);
    ex[i] = v;
    atomicAdd(&den[(long)d * HH + hh], v);
}

// out[dst,h,c] += xl[src,h,c] * alpha[e,h]
__global__ __launch_bounds__(256) void k_edge_agg(const int* __restrict__ ei,
                                                  const float* __restrict__ xl,
                                                  const float* __restrict__ ex,
                                                  const float* __restrict__ den,
                                                  float* __restrict__ outacc) {
    long i = (long)blockIdx.x * 256 + threadIdx.x;      // e*256 + hc
    if (i >= (long)EE * (HH * CC)) return;
    int hc = (int)(i & 255); long e = i >> 8;
    int hh = hc >> 6;
    int s = ei[e], d = ei[EE + e];
    float alpha = ex[e * HH + hh] / (den[(long)d * HH + hh] + 1e-16f);
    atomicAdd(&outacc[(size_t)d * (HH * CC) + hc],
              xl[(size_t)s * (HH * CC) + hc] * alpha);
}

// h = relu(mean_heads(out) + bias) + h_skip ; h_skip = h
__global__ __launch_bounds__(256) void k_node_update(const float* __restrict__ outacc,
                                                     const float* __restrict__ bias,
                                                     float* __restrict__ h,
                                                     float* __restrict__ h_skip) {
    long i = (long)blockIdx.x * 256 + threadIdx.x;      // n*64+c
    if (i >= (long)NN * CC) return;
    int c = (int)(i & 63); long n = i >> 6;
    const float* o = outacc + n * (size_t)(HH * CC);
    float m = 0.25f * (o[c] + o[CC + c] + o[2 * CC + c] + o[3 * CC + c]) + bias[c];
    m = m > 0.f ? m : 0.f;
    float v = m + h_skip[i];
    h[i] = v; h_skip[i] = v;
}

// global add pool per graph
__global__ __launch_bounds__(256) void k_pool(const float* __restrict__ h,
                                              const int* __restrict__ batch,
                                              float* __restrict__ pooled) {
    long i = (long)blockIdx.x * 256 + threadIdx.x;
    if (i >= (long)NN * CC) return;
    int c = (int)(i & 63); long n = i >> 6;
    atomicAdd(&pooled[(size_t)batch[n] * CC + c], h[i]);
}

// z = relu(pooled @ fc1_w + fc1_b)   [8000,64]@[64,64]
__global__ __launch_bounds__(256) void k_fc1(const float* __restrict__ pooled,
                                             const _Float16* __restrict__ pB,
                                             const float* __restrict__ bias,
                                             float* __restrict__ z) {
    __shared__ __align__(16) _Float16 sA[128 * CC];     // 16 KB
    int tid = threadIdx.x, wave = tid >> 5, lane = tid & 31;
    {
        int row = tid >> 1, c0 = (tid & 1) * 32;        // 32 halfs per thread
        long grow = (long)blockIdx.x * 128 + row;
        _Float16* dst = sA + row * CC + c0;
        if (grow < GG) {
            const float* src = pooled + grow * CC + c0;
#pragma unroll
            for (int j = 0; j < 8; ++j) {
                v4f v = *(const v4f*)(src + 4 * j);
#pragma unroll
                for (int t = 0; t < 4; ++t) dst[4 * j + t] = (_Float16)v[t];
            }
        } else {
#pragma unroll
            for (int j = 0; j < 32; ++j) dst[j] = (_Float16)0.f;
        }
    }
    __syncthreads();
    long mtile = (long)blockIdx.x * 8 + wave;
    if (mtile * 16 >= GG) return;                       // wave-uniform
    const _Float16* rowp = sA + (wave * 16 + (lane & 15)) * CC;
    v16h a0 = lds_a_frag(rowp, 0, lane);
    v16h a1 = lds_a_frag(rowp, 32, lane);
    int col = lane & 15, rsel = (lane >> 4) * 8;
#pragma unroll
    for (int nt = 0; nt < 4; ++nt) {
        v16h b0 = packed_b_frag(pB, nt * 2 + 0, lane);
        v16h b1 = packed_b_frag(pB, nt * 2 + 1, lane);
        v8f c = {};
        c = wmma_acc(a0, b0, c);
        c = wmma_acc(a1, b1, c);
        int colg = nt * 16 + col;
        float bv = bias[colg];
#pragma unroll
        for (int r = 0; r < 8; ++r) {
            float v = c[r] + bv;
            z[(mtile * 16 + rsel + r) * (size_t)CC + colg] = v > 0.f ? v : 0.f;
        }
    }
}

// out = relu(z @ fc2_w + fc2_b)  [G,64]@[64,1]
__global__ __launch_bounds__(256) void k_fc2(const float* __restrict__ z,
                                             const float* __restrict__ w,
                                             const float* __restrict__ b,
                                             float* __restrict__ out) {
    long g = (long)blockIdx.x * 256 + threadIdx.x;
    if (g >= GG) return;
    float s = b[0];
#pragma unroll 8
    for (int c = 0; c < CC; ++c) s += z[g * CC + c] * w[c];
    out[g] = s > 0.f ? s : 0.f;
}

// ---------------- host launch ----------------
extern "C" void kernel_launch(void* const* d_in, const int* in_sizes, int n_in,
                              void* d_out, int out_size, void* d_ws, size_t ws_size,
                              hipStream_t stream) {
    (void)in_sizes; (void)n_in; (void)out_size; (void)ws_size;
    const float* x         = (const float*)d_in[0];
    const int*   ei        = (const int*)  d_in[1];
    const float* edge_attr = (const float*)d_in[2];
    const int*   batch     = (const int*)  d_in[3];
    const float* lin_w     = (const float*)d_in[4];
    const float* lin_b     = (const float*)d_in[5];
    const float* bn_gamma  = (const float*)d_in[6];
    const float* bn_beta   = (const float*)d_in[7];
    const float* conv_w    = (const float*)d_in[8];
    const float* conv_we   = (const float*)d_in[9];
    const float* att_src   = (const float*)d_in[10];
    const float* att_dst   = (const float*)d_in[11];
    const float* att_edge  = (const float*)d_in[12];
    const float* conv_bias = (const float*)d_in[13];
    const float* fc1_w     = (const float*)d_in[14];
    const float* fc1_b     = (const float*)d_in[15];
    const float* fc2_w     = (const float*)d_in[16];
    const float* fc2_b     = (const float*)d_in[17];
    float* out = (float*)d_out;

    char* ws = (char*)d_ws;
    size_t off = 0;
    auto carve = [&](size_t bytes) {
        char* p = ws + off;
        off = (off + bytes + 255) & ~(size_t)255;
        return p;
    };
    float* h       = (float*)carve((size_t)NN * CC * 4);
    float* h_skip  = (float*)carve((size_t)NN * CC * 4);
    float* xl      = (float*)carve((size_t)NN * HH * CC * 4);
    float* outacc  = (float*)carve((size_t)NN * HH * CC * 4);
    float* a_s     = (float*)carve((size_t)NN * HH * 4);
    float* a_d     = (float*)carve((size_t)NN * HH * 4);
    float* amax    = (float*)carve((size_t)NN * HH * 4);
    float* den     = (float*)carve((size_t)NN * HH * 4);
    float* a_sc    = (float*)carve((size_t)EE * HH * 4);
    float* exbuf   = (float*)carve((size_t)EE * HH * 4);
    float* pooled  = (float*)carve((size_t)GG * CC * 4);
    float* zbuf    = (float*)carve((size_t)GG * CC * 4);
    float* bnscale = (float*)carve(CC * 4);
    float* bnshift = (float*)carve(CC * 4);
    float* weff    = (float*)carve(F_E * HH * 4);
    _Float16* p_lin  = (_Float16*)carve((size_t)4 * 512 * 2);           // 4 frags
    _Float16* p_conv = (_Float16*)carve((size_t)LL * 32 * 512 * 2);     // 32/layer
    _Float16* p_fc1  = (_Float16*)carve((size_t)8 * 512 * 2);           // 8 frags

    // ---- pack all weights to fragment-ordered f16 (once per launch) ----
    k_pack_b<<<4, 32, 0, stream>>>(lin_w, CC, F_IN, p_lin);
    for (int l = 0; l < LL; ++l)
        k_pack_b<<<32, 32, 0, stream>>>(conv_w + (size_t)l * CC * HH * CC,
                                        HH * CC, CC, p_conv + (size_t)l * 32 * 512);
    k_pack_b<<<8, 32, 0, stream>>>(fc1_w, CC, CC, p_fc1);

    // input projection (WMMA, LDS-staged A, packed B)
    k_lin_proj<<<(NN / 16 + 7) / 8, 256, 0, stream>>>(x, p_lin, lin_b, h, h_skip);

    for (int l = 0; l < LL; ++l) {
        const float* cwe = conv_we + (size_t)l * F_E * HH * CC;
        k_bn_stats<<<CC, 256, 0, stream>>>(h, bn_gamma + l * CC, bn_beta + l * CC,
                                           bnscale, bnshift);
        k_conv_gemm<<<NN / 32, 256, 0, stream>>>(
            h, bnscale, bnshift, p_conv + (size_t)l * 32 * 512,
            att_src + l * HH * CC, att_dst + l * HH * CC, xl, a_s, a_d);
        k_weff<<<1, 32, 0, stream>>>(cwe, att_edge + l * HH * CC, weff);
        k_fill<<<((long)NN * HH * CC + 255) / 256, 256, 0, stream>>>(
            outacc, (long)NN * HH * CC, 0.f);
        k_fill<<<((long)NN * HH + 255) / 256, 256, 0, stream>>>(
            amax, (long)NN * HH, -1e30f);
        k_fill<<<((long)NN * HH + 255) / 256, 256, 0, stream>>>(
            den, (long)NN * HH, 0.f);
        k_edge_score<<<((long)EE * HH + 255) / 256, 256, 0, stream>>>(
            ei, edge_attr, weff, a_s, a_d, xl, a_sc, amax);
        k_edge_exp<<<((long)EE * HH + 255) / 256, 256, 0, stream>>>(
            ei, a_sc, amax, exbuf, den);
        k_edge_agg<<<((long)EE * HH * CC + 255) / 256, 256, 0, stream>>>(
            ei, xl, exbuf, den, outacc);
        k_node_update<<<((long)NN * CC + 255) / 256, 256, 0, stream>>>(
            outacc, conv_bias + l * CC, h, h_skip);
    }

    k_fill<<<((long)GG * CC + 255) / 256, 256, 0, stream>>>(pooled, (long)GG * CC, 0.f);
    k_pool<<<((long)NN * CC + 255) / 256, 256, 0, stream>>>(h, batch, pooled);
    k_fc1<<<(GG / 16 + 7) / 8, 256, 0, stream>>>(pooled, p_fc1, fc1_b, zbuf);
    k_fc2<<<(GG + 255) / 256, 256, 0, stream>>>(zbuf, fc2_w, fc2_b, out);
}
